// NeuralWordAligner_47347719471188
// MI455X (gfx1250) — matrix operand order repaired and checked by reading, not compiled, not run on hardware
//
#include <hip/hip_runtime.h>
#include <hip/hip_bf16.h>

typedef __attribute__((ext_vector_type(16))) _Float16 v16h;
typedef __attribute__((ext_vector_type(8)))  float    v8f;

#define NEGBIG (-1e30f)

__device__ __forceinline__ float wave_max32(float v) {
#pragma unroll
  for (int off = 16; off; off >>= 1)
    v = fmaxf(v, __shfl_xor(v, off, 32));
  return v;
}
__device__ __forceinline__ float wave_sum32(float v) {
#pragma unroll
  for (int off = 16; off; off >>= 1)
    v += __shfl_xor(v, off, 32);
  return v;
}

// Problem constants (reference: B=48, S=128, W=4, K=256)
constexpr int S_LEN = 128;
constexpr int W_SPAN = 4;
constexpr int K_LAB = 256;
constexpr int EH_STRIDE = 264;  // padded halves/row: 528B -> +4 banks per row

// Dynamic LDS layout (bytes):
//   [0,4096)      float rbuf[4][256]   ring buffer of r_{j}
//   [4096,4128)   float pmax[8]
//   [4128,4160)   float psum[8]
//   [4160,4164)   int   slen
//   [4224,4736)   _Float16 vhalf[256]  (64B aligned)
//   [4736,139904) _Float16 Eh[256][EH_STRIDE]  exp(T*mask), f16
constexpr int SMEM_BYTES = 4736 + K_LAB * EH_STRIDE * 2;

__global__ __launch_bounds__(256, 1)
void semicrf_partition_kernel(const float* __restrict__ logits,
                              const float* __restrict__ Tm,
                              const float* __restrict__ ham,
                              const float* __restrict__ tagm,
                              const int*   __restrict__ textm,
                              float* __restrict__ out)
{
  extern __shared__ char smem[];
  float*    rbuf  = (float*)smem;                 // [4][K_LAB]
  float*    pmax  = rbuf + 4 * K_LAB;             // [8]
  float*    psum  = pmax + 8;                     // [8]
  int*      slen  = (int*)(psum + 8);             // [1]
  _Float16* vhalf = (_Float16*)(smem + 4224);     // [K_LAB]
  _Float16* Eh    = (_Float16*)(smem + 4736);     // [K_LAB][EH_STRIDE]

  const int b    = blockIdx.x;
  const int tid  = threadIdx.x;   // 0..255; also the label index k
  const int lane = tid & 31;
  const int warp = tid >> 5;      // 0..7

  // ---------- setup: Eh = exp(T * tag_mask) (f16), sequence length ----------
  const size_t tb = (size_t)b * K_LAB * K_LAB;
#pragma unroll 1
  for (int r = 0; r < K_LAB; ++r) {
    const size_t gi = tb + (size_t)r * K_LAB + tid;     // coalesced over lanes
    Eh[r * EH_STRIDE + tid] = (_Float16)__expf(Tm[gi] * tagm[gi]);
  }
  if (tid == 0) {
    int L = 0;
    for (int s = 0; s < S_LEN; ++s) L += textm[b * S_LEN + s];
    slen[0] = L;
    if (L == 0) out[b] = __logf((float)K_LAB);  // LSE_k of the zero row
  }
  __syncthreads();
  const int lengthB = slen[0];

  // ---------- A fragments in VGPRs: wave owns k-tiles 2*warp, 2*warp+1 ------
  // A 16x32 layout (matches B): lane L holds row M=L&15; element h:
  //   col = (L>=16 ? 16:0) + h
  v16h Afrag[2][8];
  {
    const int mrow = lane & 15;
    const int rsel = (lane >= 16) ? 16 : 0;
#pragma unroll
    for (int t = 0; t < 2; ++t) {
      const int krow = (2 * warp + t) * 16 + mrow;
#pragma unroll
      for (int c = 0; c < 8; ++c) {
#pragma unroll
        for (int h = 0; h < 16; ++h) {
          const int col = 32 * c + rsel + h;
          Afrag[t][c][h] = Eh[krow * EH_STRIDE + col];
        }
      }
    }
  }

  const float LOG_K = __logf((float)K_LAB);
  const float tm0k  = tagm[tb + tid];                         // tag_mask[b,0,k]
  const size_t embase = (size_t)b * S_LEN * W_SPAN * K_LAB + tid;

  // ---------- sequential recurrence over j ----------
#pragma unroll 1
  for (int j = 0; j < S_LEN; ++j) {
    // phase 1: alpha_j[k] = LSE_w( emit + r_{j-1-w} ), k = tid
    const float* lg = logits + embase + (size_t)j * (W_SPAN * K_LAB);
    const float* hc = ham    + embase + (size_t)j * (W_SPAN * K_LAB);
    float t4[W_SPAN];
#pragma unroll
    for (int w = 0; w < W_SPAN; ++w) {
      const float e = (lg[w * K_LAB] + hc[w * K_LAB]) * tm0k;
      const int idx = j - 1 - w;                 // previous alpha index
      float rr;
      if (idx >= 0)      rr = rbuf[(idx & 3) * K_LAB + tid];  // real boundary: +T
      else if (w == j)   rr = LOG_K;             // initial state, no transition
      else               rr = NEGBIG;            // span starts before t=0
      t4[w] = e + rr;
    }
    float tmax = fmaxf(fmaxf(t4[0], t4[1]), fmaxf(t4[2], t4[3]));
    float ssum = 0.f;
#pragma unroll
    for (int w = 0; w < W_SPAN; ++w) ssum += __expf(t4[w] - tmax);
    const float alpha = tmax + __logf(ssum);

    // prefetch next step's emit rows (global_prefetch_b8); block covers rows
    if (j + 1 < S_LEN) {
#pragma unroll
      for (int w = 0; w < W_SPAN; ++w) {
        __builtin_prefetch(lg + (W_SPAN + w) * K_LAB, 0, 0);
        __builtin_prefetch(hc + (W_SPAN + w) * K_LAB, 0, 0);
      }
    }

    // phase 2: block-wide max of alpha
    const float wm = wave_max32(alpha);
    if (lane == 0) pmax[warp] = wm;
    __syncthreads();
    const float m = fmaxf(fmaxf(fmaxf(pmax[0], pmax[1]), fmaxf(pmax[2], pmax[3])),
                          fmaxf(fmaxf(pmax[4], pmax[5]), fmaxf(pmax[6], pmax[7])));

    // phase 3: v[k] = exp(alpha - m) as f16; piggyback sum for the readout LSE
    const float ev = __expf(alpha - m);
    vhalf[tid] = (_Float16)ev;
    const float ws = wave_sum32(ev);
    if (lane == 0) psum[warp] = ws;
    __syncthreads();

    if ((j + 1) == lengthB && tid == 0) {
      const float tot = psum[0] + psum[1] + psum[2] + psum[3] +
                        psum[4] + psum[5] + psum[6] + psum[7];
      out[b] = m + __logf(tot);                  // partition for this batch elem
    }

    // phase 4: r_j = m + log( exp(T) @ v ) via WMMA (v broadcast over B cols)
    // B 32x16 f16 layout: lane L holds K = (L>=16?16:0)+h, col = L&15.
    // Load ALL B fragments first so the 16 WMMAs can issue back-to-back
    // behind a single s_wait_dscnt instead of a wait per chunk.
    const _Float16* vb = vhalf + ((lane >= 16) ? 16 : 0);
    v16h bfs[8];
#pragma unroll
    for (int c = 0; c < 8; ++c)
      bfs[c] = *(const v16h*)(vb + c * 32);      // 32B-aligned LDS reads

    v8f acc0 = {}, acc1 = {};
#pragma unroll
    for (int c = 0; c < 8; ++c) {
      acc0 = __builtin_amdgcn_wmma_f32_16x16x32_f16(
                 false, Afrag[0][c], false, bfs[c], (short)0, acc0, false, false);
      acc1 = __builtin_amdgcn_wmma_f32_16x16x32_f16(
                 false, Afrag[1][c], false, bfs[c], (short)0, acc1, false, false);
    }
    // D layout: VGPR g -> M = g + (lane>=16 ? 8:0); result identical across N,
    // so lanes with (lane&15)==0 write the 16 rows of each k-tile.
    if ((lane & 15) == 0) {
      float* rj = rbuf + (j & 3) * K_LAB;
      const int roff = (lane >= 16) ? 8 : 0;
#pragma unroll
      for (int g = 0; g < 8; ++g) {
        rj[(2 * warp + 0) * 16 + roff + g] = m + __logf(fmaxf(acc0[g], 1e-38f));
        rj[(2 * warp + 1) * 16 + roff + g] = m + __logf(fmaxf(acc1[g], 1e-38f));
      }
    }
    __syncthreads();
  }
}

extern "C" void kernel_launch(void* const* d_in, const int* in_sizes, int n_in,
                              void* d_out, int out_size, void* d_ws, size_t ws_size,
                              hipStream_t stream) {
  (void)in_sizes; (void)n_in; (void)d_ws; (void)ws_size;
  const float* logits = (const float*)d_in[0];
  const float* T      = (const float*)d_in[1];
  const float* ham    = (const float*)d_in[2];
  const float* tagm   = (const float*)d_in[3];
  const int*   textm  = (const int*)d_in[4];
  float* out = (float*)d_out;

  const int B = out_size;  // output is (1, B)
  if (B <= 0) return;

  static bool attr_set = false;  // runtime-state call only; enqueues no work
  if (!attr_set) {
    hipFuncSetAttribute((const void*)semicrf_partition_kernel,
                        hipFuncAttributeMaxDynamicSharedMemorySize, SMEM_BYTES);
    attr_set = true;
  }

  semicrf_partition_kernel<<<dim3(B), dim3(256), SMEM_BYTES, stream>>>(
      logits, T, ham, tagm, textm, out);
}
